// Plamo2MambaDecoderLayer_19464791785817
// MI455X (gfx1250) — matrix operand order repaired and checked by reading, not compiled, not run on hardware
//
#include <hip/hip_runtime.h>
#include <stdint.h>

#define T_LEN 1024
#define H_DIM 2048
#define NH_ 32
#define HPH_ 128
#define N_ 64
#define DT_RANK_ 128
#define I_ 4096
#define IM_ 8192

#define SCAN_ASYNC 1

typedef __attribute__((ext_vector_type(16))) __bf16 v16bf;
typedef __attribute__((ext_vector_type(8))) float v8f;
typedef __attribute__((ext_vector_type(4))) float v4f;
typedef __attribute__((ext_vector_type(2))) float v2f;
typedef __attribute__((ext_vector_type(4))) unsigned int v4u;
typedef __attribute__((ext_vector_type(2))) unsigned int v2u;

union FragU { v4u q[2]; v16bf v; };

// ---------------- GEMM: C[M,N] = A[M,K] @ W[N,K]^T  (fp32 in/out, bf16 WMMA) ---
// 128x128 tile per 256-thread block (8 waves, 2x4), 8 WMMA (16x16x32) per wave
// per K-step, double-buffered LDS, one barrier per K-step.
#define TM 128
#define TN 128
#define TKK 32
#define PA 40    // lA pitch (bf16 elems): 32 + 8 pad, 80B rows (16B aligned)
#define PB 136   // lBt pitch (bf16 elems): 128 + 8 pad, 272B rows (16B aligned)
#define PBW 68   // lBt pitch in u32

__device__ __forceinline__ unsigned fu(float x) { return __float_as_uint(x); }
// pack {bf16(lo), bf16(hi)} (truncation) in ONE v_perm_b32
__device__ __forceinline__ unsigned pkhi(unsigned lo_src, unsigned hi_src) {
  return __builtin_amdgcn_perm(hi_src, lo_src, 0x07060302u);
}

template <int K>
__device__ __forceinline__ void load_tileA(const float* __restrict__ A, int m0,
                                           int k0, int tid, v4f ar[4]) {
#pragma unroll
  for (int it = 0; it < 4; ++it) {
    int q = tid + it * 256;        // 1024 float4 units: 128 rows x 8
    int row = q >> 3, c4 = q & 7;
    ar[it] = *(const v4f*)(A + (size_t)(m0 + row) * K + k0 + c4 * 4);
  }
}
__device__ __forceinline__ void store_tileA(unsigned short* lA, int tid,
                                            const v4f ar[4]) {
#pragma unroll
  for (int it = 0; it < 4; ++it) {
    int q = tid + it * 256;
    int row = q >> 3, c4 = q & 7;
    v2u p;
    p.x = pkhi(fu(ar[it].x), fu(ar[it].y));
    p.y = pkhi(fu(ar[it].z), fu(ar[it].w));
    *(v2u*)&lA[row * PA + c4 * 4] = p;
  }
}
template <int K>
__device__ __forceinline__ void load_tileB(const float* __restrict__ W, int n0,
                                           int k0, int tid, v2f br0[4], v2f br1[4]) {
#pragma unroll
  for (int it = 0; it < 4; ++it) {
    int q = tid + it * 256;        // 1024 units: 64 row-pairs x 16 k-pairs
    int npair = q >> 4, kpair = q & 15;
    const float* base = W + (size_t)(n0 + 2 * npair) * K + k0 + kpair * 2;
    br0[it] = *(const v2f*)base;           // row n0+2*npair
    br1[it] = *(const v2f*)(base + K);     // row n0+2*npair+1
  }
}
__device__ __forceinline__ void store_tileB(unsigned short* lBt, int tid,
                                            const v2f br0[4], const v2f br1[4]) {
  unsigned* l32 = (unsigned*)lBt;  // lBt[k][n] as u32 pairs of n
#pragma unroll
  for (int it = 0; it < 4; ++it) {
    int q = tid + it * 256;
    int npair = q >> 4, kpair = q & 15;
    // u32 = { bf16(row n0 @ k), bf16(row n1 @ k) } = n-pair at fixed k
    l32[(2 * kpair + 0) * PBW + npair] = pkhi(fu(br0[it].x), fu(br1[it].x));
    l32[(2 * kpair + 1) * PBW + npair] = pkhi(fu(br0[it].y), fu(br1[it].y));
  }
}

template <int N, int K>
__global__ __launch_bounds__(256) void gemm_nt_wmma(
    const float* __restrict__ A, const float* __restrict__ W,
    float* __restrict__ C) {
  __shared__ unsigned short lA[2][TM * PA];    // A tile, row-major [m][k] bf16
  __shared__ unsigned short lBt[2][TKK * PB];  // W tile transposed [k][n] bf16
  const int tid = threadIdx.x;
  const int lane = tid & 31, wid = tid >> 5;
  const int m0 = blockIdx.y * TM, n0 = blockIdx.x * TN;
  const int wm = (wid >> 2) * 64;  // 2 wave-rows
  const int wn = (wid & 3) * 32;   // 4 wave-cols

  v8f acc[4][2];
#pragma unroll
  for (int i = 0; i < 4; ++i)
#pragma unroll
    for (int j = 0; j < 2; ++j)
#pragma unroll
      for (int e = 0; e < 8; ++e) acc[i][j][e] = 0.f;

  // prologue: stage tile k0=0 into buffer 0
  v4f ar[4];
  v2f br0[4], br1[4];
  load_tileA<K>(A, m0, 0, tid, ar);
  load_tileB<K>(W, n0, 0, tid, br0, br1);
  store_tileA(lA[0], tid, ar);
  store_tileB(lBt[0], tid, br0, br1);
  __syncthreads();

  int cur = 0;
  for (int k0 = 0; k0 < K; k0 += TKK, cur ^= 1) {
    const bool hasNext = (k0 + TKK) < K;
    if (hasNext) {  // prefetch next tile into regs (latency hides under WMMA)
      load_tileA<K>(A, m0, k0 + TKK, tid, ar);
      load_tileB<K>(W, n0, k0 + TKK, tid, br0, br1);
    }
    // ---- fragments from current buffer ----
    FragU af[4], bfr[2];
#pragma unroll
    for (int i = 0; i < 4; ++i) {
      // A: lane<16 -> row M=lane, K {0..7,16..23}; lane>=16 -> K {8..15,24..31}
      int r = wm + i * 16 + (lane & 15);
      int kk = (lane & 16) ? 8 : 0;
      const unsigned short* p = &lA[cur][r * PA + kk];
      af[i].q[0] = *(const v4u*)p;
      af[i].q[1] = *(const v4u*)(p + 16);
    }
#pragma unroll
    for (int j = 0; j < 2; ++j) {
      // B: lane<16 -> col N=lane, K {0..7,16..23}; lane>=16 -> K {8..15,24..31}
      const unsigned short* p = &lBt[cur][((lane & 16) ? 8 : 0) * PB + wn + j * 16 + (lane & 15)];
      unsigned short* e = (unsigned short*)&bfr[j];
#pragma unroll
      for (int s = 0; s < 8; ++s) {
        e[s] = p[s * PB];
        e[8 + s] = p[(16 + s) * PB];
      }
    }
#pragma unroll
    for (int i = 0; i < 4; ++i)
#pragma unroll
      for (int j = 0; j < 2; ++j)
        acc[i][j] = __builtin_amdgcn_wmma_f32_16x16x32_bf16(
            false, af[i].v, false, bfr[j].v, (short)0, acc[i][j], false, false);
    if (hasNext) {  // convert + store next tile into the other buffer
      store_tileA(lA[cur ^ 1], tid, ar);
      store_tileB(lBt[cur ^ 1], tid, br0, br1);
    }
    __syncthreads();
  }
  // ---- epilogue: C/D layout: VGPR d -> M=d (lanes<16) / M=8+d, lanes -> N ----
  const int nc = n0 + wn + (lane & 15);
  const int mb = m0 + wm + ((lane & 16) ? 8 : 0);
  float* cb = C + (size_t)mb * N + nc;
#pragma unroll
  for (int i = 0; i < 4; ++i)
#pragma unroll
    for (int j = 0; j < 2; ++j)
#pragma unroll
      for (int d = 0; d < 8; ++d)
        cb[(size_t)(i * 16 + d) * N + j * 16] = acc[i][j][d];
}

// ---------------- RMSNorm (optionally strided slice, optional residual) -------
__device__ __forceinline__ float block_sum(float v) {
  __shared__ float red[8];
  int lane = threadIdx.x & 31, wid = threadIdx.x >> 5;
#pragma unroll
  for (int off = 16; off > 0; off >>= 1) v += __shfl_xor(v, off, 32);
  if (lane == 0) red[wid] = v;
  __syncthreads();
  float s = (threadIdx.x < 8) ? red[threadIdx.x] : 0.f;
#pragma unroll
  for (int off = 4; off > 0; off >>= 1) s += __shfl_xor(s, off, 32);
  if (threadIdx.x == 0) red[0] = s;
  __syncthreads();
  return red[0];
}

__global__ __launch_bounds__(256) void rmsnorm_kernel(
    const float* __restrict__ x, int rstride, int coff, int cols,
    const float* __restrict__ w, const float* __restrict__ resid,
    float* __restrict__ out, int ostride) {
  const float* row = x + (size_t)blockIdx.x * rstride + coff;
  float ss = 0.f;
  for (int c = threadIdx.x; c < cols; c += 256) { float v = row[c]; ss += v * v; }
  ss = block_sum(ss);
  float sc = rsqrtf(ss / (float)cols + 1e-6f);
  float* orow = out + (size_t)blockIdx.x * ostride;
  if (resid) {
    const float* rr = resid + (size_t)blockIdx.x * ostride;
    for (int c = threadIdx.x; c < cols; c += 256) orow[c] = rr[c] + row[c] * sc * w[c];
  } else {
    for (int c = threadIdx.x; c < cols; c += 256) orow[c] = row[c] * sc * w[c];
  }
}

// ---------------- depthwise causal conv (K=4) + silu --------------------------
__global__ __launch_bounds__(256) void conv_silu_kernel(
    const float* __restrict__ proj, const float* __restrict__ cw,
    float* __restrict__ xc) {
  int i = blockIdx.x * 256 + threadIdx.x;
  int t = blockIdx.y;
  const float* base = proj + I_ + i;  // x half of proj, column i, row stride 2I
  float acc = base[(size_t)t * (2 * I_)] * cw[i * 4 + 3];
  if (t >= 1) acc += base[(size_t)(t - 1) * (2 * I_)] * cw[i * 4 + 2];
  if (t >= 2) acc += base[(size_t)(t - 2) * (2 * I_)] * cw[i * 4 + 1];
  if (t >= 3) acc += base[(size_t)(t - 3) * (2 * I_)] * cw[i * 4 + 0];
  xc[(size_t)t * I_ + i] = acc / (1.f + __expf(-acc));
}

// ---------------- dt = tsn @ dt_proj_w^T + b -> softplus ----------------------
__global__ __launch_bounds__(256) void dt_softplus_kernel(
    const float* __restrict__ tsn, const float* __restrict__ w,
    const float* __restrict__ b, float* __restrict__ delta) {
  int idx = blockIdx.x * 256 + threadIdx.x;  // T*NH
  int t = idx >> 5, nh = idx & 31;
  const float* tr = tsn + (size_t)t * DT_RANK_;
  const float* wr = w + (size_t)nh * DT_RANK_;
  float acc = 0.f;
#pragma unroll 8
  for (int r = 0; r < DT_RANK_; ++r) acc += tr[r] * wr[r];
  float v = acc + b[nh];
  delta[idx] = (v > 20.f) ? v : log1pf(__expf(v));
}

// ---------------- selective scan ---------------------------------------------
// block = 1 head (128 channels), 256 threads: lane pair (2k,2k+1) = channel k,
// each handles 32 of N=64 states; y reduced with shfl_xor(1).
// B/C rows for each t double-buffered into LDS via CDNA5 async global->LDS.
__global__ __launch_bounds__(256) void scan_kernel(
    const float* __restrict__ xc, const float* __restrict__ delta,
    const float* __restrict__ bc,  // Bn [T,64] followed by Cn [T,64]
    const float* __restrict__ A, float* __restrict__ ys) {
  __shared__ float sBC[2][128];  // [buf][ B:0..63 | C:64..127 ]
  const int tid = threadIdx.x;
  const int h = blockIdx.x;
  const int cl = tid >> 1;  // channel in head
  const int ng = tid & 1;   // state half
  const int c = h * HPH_ + cl;

  float a2[32], s[32];
#pragma unroll
  for (int n = 0; n < 32; ++n) {
    a2[n] = A[(size_t)c * N_ + ng * 32 + n] * 1.44269504f;  // A * log2(e)
    s[n] = 0.f;
  }
#if SCAN_ASYNC
  unsigned ldsbase = (unsigned)(size_t)(&sBC[0][0]);
  if (tid < 128) {  // prefetch t = 0 into buf 0
    unsigned voff = (tid < 64) ? (unsigned)(tid * 4)
                               : (unsigned)((T_LEN * 64 + (tid - 64)) * 4);
    unsigned loff = ldsbase + (unsigned)(tid * 4);
    asm volatile("global_load_async_to_lds_b32 %0, %1, %2"
                 :: "v"(loff), "v"(voff), "s"(bc) : "memory");
  }
#endif
  for (int t = 0; t < T_LEN; ++t) {
    const int buf = t & 1;
#if SCAN_ASYNC
    asm volatile("s_wait_asynccnt 0" ::: "memory");
    __syncthreads();  // sBC[buf] ready; everyone done with sBC[buf^1]
    if (t + 1 < T_LEN && tid < 128) {
      int tn = t + 1;
      unsigned voff = (tid < 64)
          ? (unsigned)((tn * 64 + tid) * 4)
          : (unsigned)((T_LEN * 64 + tn * 64 + (tid - 64)) * 4);
      unsigned loff = ldsbase + (unsigned)(((buf ^ 1) * 128 + tid) * 4);
      asm volatile("global_load_async_to_lds_b32 %0, %1, %2"
                   :: "v"(loff), "v"(voff), "s"(bc) : "memory");
    }
#else
    __syncthreads();
    if (tid < 128)
      sBC[buf][tid] = (tid < 64) ? bc[(size_t)t * 64 + tid]
                                 : bc[(size_t)T_LEN * 64 + (size_t)t * 64 + tid - 64];
    __syncthreads();
#endif
    float d = delta[t * NH_ + h];
    float dx = d * xc[(size_t)t * I_ + c];
    const float* B_ = &sBC[buf][ng * 32];
    const float* C_ = &sBC[buf][64 + ng * 32];
    float y = 0.f;
#pragma unroll
    for (int n = 0; n < 32; ++n) {
      float dA = __builtin_amdgcn_exp2f(d * a2[n]);  // exp(delta*A)
      s[n] = s[n] * dA + dx * B_[n];
      y += s[n] * C_[n];
    }
    y += __shfl_xor(y, 1, 32);
    if (ng == 0) ys[(size_t)t * I_ + c] = y;
  }
}

// ---------------- y = (ys + xc*D) * silu(gate) -------------------------------
__global__ __launch_bounds__(256) void gate_kernel(
    const float* __restrict__ ys, const float* __restrict__ xc,
    const float* __restrict__ D, const float* __restrict__ proj,
    float* __restrict__ yg) {
  int i = blockIdx.x * 256 + threadIdx.x;
  int t = blockIdx.y;
  float g = proj[(size_t)t * (2 * I_) + i];  // gate = first half
  float sg = g / (1.f + __expf(-g));
  size_t off = (size_t)t * I_ + i;
  yg[off] = (ys[off] + xc[off] * D[i]) * sg;
}

// ---------------- SwiGLU -----------------------------------------------------
__global__ __launch_bounds__(256) void swiglu_kernel(
    const float* __restrict__ gu, float* __restrict__ mid) {
  int i = blockIdx.x * 256 + threadIdx.x;
  int t = blockIdx.y;
  float g0 = gu[(size_t)t * (2 * IM_) + i];
  float g1 = gu[(size_t)t * (2 * IM_) + IM_ + i];
  mid[(size_t)t * IM_ + i] = g0 / (1.f + __expf(-g0)) * g1;
}

// ---------------- workspace layout (floats) ----------------------------------
constexpr size_t SZ_HN = (size_t)T_LEN * H_DIM;
constexpr size_t OF_HN = 0;                                   // also h2n (alias)
constexpr size_t OF_PROJ = OF_HN + SZ_HN;                     // also mid (alias)
constexpr size_t SZ_PROJ = (size_t)T_LEN * 2 * I_;
constexpr size_t OF_XC = OF_PROJ + SZ_PROJ;                   // also mlp_out (alias)
constexpr size_t SZ_XC = (size_t)T_LEN * I_;
constexpr size_t OF_SSM = OF_XC + SZ_XC;
constexpr size_t SZ_SSM = (size_t)T_LEN * 256;
constexpr size_t OF_BC = OF_SSM + SZ_SSM;
constexpr size_t SZ_BC = (size_t)T_LEN * 128;
constexpr size_t OF_TSN = OF_BC + SZ_BC;
constexpr size_t SZ_TSN = (size_t)T_LEN * DT_RANK_;
constexpr size_t OF_DEL = OF_TSN + SZ_TSN;
constexpr size_t SZ_DEL = (size_t)T_LEN * NH_;
constexpr size_t OF_YS = OF_DEL + SZ_DEL;                     // gated in place
constexpr size_t SZ_YS = (size_t)T_LEN * I_;
constexpr size_t OF_MIX = OF_YS + SZ_YS;
constexpr size_t OF_H1 = OF_MIX + SZ_HN;
constexpr size_t OF_GU = OF_H1 + SZ_HN;

extern "C" void kernel_launch(void* const* d_in, const int* in_sizes, int n_in,
                              void* d_out, int out_size, void* d_ws, size_t ws_size,
                              hipStream_t stream) {
  (void)in_sizes; (void)n_in; (void)out_size; (void)ws_size;
  const float* hid        = (const float*)d_in[0];
  const float* in_proj_w  = (const float*)d_in[1];
  const float* conv_w     = (const float*)d_in[2];
  const float* x_proj_w   = (const float*)d_in[3];
  const float* dt_proj_w  = (const float*)d_in[4];
  const float* dt_proj_b  = (const float*)d_in[5];
  const float* Amat       = (const float*)d_in[6];
  const float* Dvec       = (const float*)d_in[7];
  const float* out_proj_w = (const float*)d_in[8];
  const float* pre_mix_w  = (const float*)d_in[9];
  const float* post_mix_w = (const float*)d_in[10];
  const float* pre_mlp_w  = (const float*)d_in[11];
  const float* post_mlp_w = (const float*)d_in[12];
  const float* dt_ln_w    = (const float*)d_in[13];
  const float* b_ln_w     = (const float*)d_in[14];
  const float* c_ln_w     = (const float*)d_in[15];
  const float* gate_up_w  = (const float*)d_in[16];
  const float* down_w     = (const float*)d_in[17];
  float* out = (float*)d_out;
  float* ws = (float*)d_ws;

  // 1. h_norm = rmsnorm(hidden, pre_mixer_norm_w)
  rmsnorm_kernel<<<T_LEN, 256, 0, stream>>>(hid, H_DIM, 0, H_DIM, pre_mix_w,
                                            nullptr, ws + OF_HN, H_DIM);
  // 2. proj = h_norm @ in_proj_w^T        [1024, 8192]
  gemm_nt_wmma<2 * I_, H_DIM><<<dim3(2 * I_ / TN, T_LEN / TM), 256, 0, stream>>>(
      ws + OF_HN, in_proj_w, ws + OF_PROJ);
  // 3. xconv = silu(causal_conv(x part))
  conv_silu_kernel<<<dim3(I_ / 256, T_LEN), 256, 0, stream>>>(ws + OF_PROJ,
                                                              conv_w, ws + OF_XC);
  // 4. ssm = xconv @ x_proj_w^T           [1024, 256]
  gemm_nt_wmma<256, I_><<<dim3(256 / TN, T_LEN / TM), 256, 0, stream>>>(
      ws + OF_XC, x_proj_w, ws + OF_SSM);
  // 5-7. small rmsnorms: B, C, ts
  rmsnorm_kernel<<<T_LEN, 256, 0, stream>>>(ws + OF_SSM, 256, 0, N_, b_ln_w,
                                            nullptr, ws + OF_BC, N_);
  rmsnorm_kernel<<<T_LEN, 256, 0, stream>>>(ws + OF_SSM, 256, N_, N_, c_ln_w,
                                            nullptr, ws + OF_BC + (size_t)T_LEN * N_, N_);
  rmsnorm_kernel<<<T_LEN, 256, 0, stream>>>(ws + OF_SSM, 256, 2 * N_, DT_RANK_,
                                            dt_ln_w, nullptr, ws + OF_TSN, DT_RANK_);
  // 8. delta = softplus(tsn @ dt_proj_w^T + b)
  dt_softplus_kernel<<<(T_LEN * NH_) / 256, 256, 0, stream>>>(
      ws + OF_TSN, dt_proj_w, dt_proj_b, ws + OF_DEL);
  // 9. selective scan -> ys
  scan_kernel<<<NH_, 256, 0, stream>>>(ws + OF_XC, ws + OF_DEL, ws + OF_BC,
                                       Amat, ws + OF_YS);
  // 10. ys = (ys + xc*D) * silu(gate)   (in place)
  gate_kernel<<<dim3(I_ / 256, T_LEN), 256, 0, stream>>>(
      ws + OF_YS, ws + OF_XC, Dvec, ws + OF_PROJ, ws + OF_YS);
  // 11. mix = ys @ out_proj_w^T           [1024, 2048]
  gemm_nt_wmma<H_DIM, I_><<<dim3(H_DIM / TN, T_LEN / TM), 256, 0, stream>>>(
      ws + OF_YS, out_proj_w, ws + OF_MIX);
  // 12. h1 = hidden + rmsnorm(mix, post_mixer_norm_w)
  rmsnorm_kernel<<<T_LEN, 256, 0, stream>>>(ws + OF_MIX, H_DIM, 0, H_DIM,
                                            post_mix_w, hid, ws + OF_H1, H_DIM);
  // 13. h2n = rmsnorm(h1, pre_mlp_norm_w)   (alias over h_norm)
  rmsnorm_kernel<<<T_LEN, 256, 0, stream>>>(ws + OF_H1, H_DIM, 0, H_DIM,
                                            pre_mlp_w, nullptr, ws + OF_HN, H_DIM);
  // 14. gu = h2n @ gate_up_w^T            [1024, 16384]
  gemm_nt_wmma<2 * IM_, H_DIM><<<dim3(2 * IM_ / TN, T_LEN / TM), 256, 0, stream>>>(
      ws + OF_HN, gate_up_w, ws + OF_GU);
  // 15. mid = silu(g0)*g1                  (alias over proj)
  swiglu_kernel<<<dim3(IM_ / 256, T_LEN), 256, 0, stream>>>(ws + OF_GU,
                                                            ws + OF_PROJ);
  // 16. mlp = mid @ down_w^T              [1024, 2048]  (alias over xconv)
  gemm_nt_wmma<H_DIM, IM_><<<dim3(H_DIM / TN, T_LEN / TM), 256, 0, stream>>>(
      ws + OF_PROJ, down_w, ws + OF_XC);
  // 17. out = h1 + rmsnorm(mlp, post_mlp_norm_w)
  rmsnorm_kernel<<<T_LEN, 256, 0, stream>>>(ws + OF_XC, H_DIM, 0, H_DIM,
                                            post_mlp_w, ws + OF_H1, out, H_DIM);
}
